// MultiHeadAttention_73950746903066
// MI455X (gfx1250) — compile-verified, matrix operand order checked by reading, not run on hardware
//
#include <hip/hip_runtime.h>

typedef __bf16 bf16;
typedef __attribute__((ext_vector_type(16))) __bf16 v16bf;
typedef __attribute__((ext_vector_type(8)))  __bf16 v8bf;
typedef __attribute__((ext_vector_type(8)))  float  v8f;

#define TPB 256

// ---------------------------------------------------------------------------
// fp32 -> bf16 bulk convert (4 elements / thread, 16B global loads)
// ---------------------------------------------------------------------------
__global__ __launch_bounds__(TPB) void cvt_f32_bf16(const float* __restrict__ in,
                                                    bf16* __restrict__ out,
                                                    long n4) {
  long i = (long)blockIdx.x * TPB + threadIdx.x;
  if (i >= n4) return;
  float4 v = ((const float4*)in)[i];
  union { bf16 b[4]; unsigned long long u; } o;
  o.b[0] = (bf16)v.x; o.b[1] = (bf16)v.y; o.b[2] = (bf16)v.z; o.b[3] = (bf16)v.w;
  ((unsigned long long*)out)[i] = o.u;
}

// ---------------------------------------------------------------------------
// Tiled WMMA GEMM:  C[M,N] = (A[M,K] @ B[N,K]^T) * scale + bias[N]
//   A: row-major [M,K] leading dim lda;  B: row-major [N,K] leading dim ldb
//   CT=false: C stored [M,N] ld ldc (f32 Cf and/or bf16 Cb)
//   CT=true : bf16 C stored transposed [N,M] ld ldc (one b128 per fragment)
// Block tile 128x128x64, 8 wave32 waves (2x4), each wave 64x32 (4x2 WMMA).
// Register-prefetch pipeline with INDIVIDUAL named int4 registers (no arrays,
// no lambdas): arrays captured by reference defeated SROA and forced the
// prefetch buffers into scratch memory (observed scratch_load_b128 in rounds
// 2-3). Scalars cannot be address-taken, so they must live in VGPRs.
// ---------------------------------------------------------------------------
template <bool CT>
__global__ __launch_bounds__(TPB, 1) void gemm_bf16_wmma(
    const bf16* __restrict__ A, const bf16* __restrict__ B,
    const float* __restrict__ bias, float scale,
    float* __restrict__ Cf, bf16* __restrict__ Cb,
    int M, int N, int K, int lda, int ldb, int ldc,
    long sA, long sB, long sC) {
  constexpr int BM = 128, BN = 128, BK = 64, LDK = BK + 8;  // 144B LDS rows
  __shared__ bf16 As[BM][LDK];
  __shared__ bf16 Bs[BN][LDK];

  const int tid  = threadIdx.x;
  const int lane = tid & 31;
  const int wave = tid >> 5;
  const int wm = wave >> 2;   // 0..1
  const int wn = wave & 3;    // 0..3
  const int g  = lane >> 4;   // K sub-block select per ISA 16-bit A layout
  const int r  = lane & 15;

  const int m0 = blockIdx.y * BM;
  const int n0 = blockIdx.x * BN;
  A += (long)blockIdx.z * sA + (size_t)m0 * lda;
  B += (long)blockIdx.z * sB;
  const long cbase = (long)blockIdx.z * sC;

  // staging geometry: 128B rows = 8 x b128 chunks; 4 row-groups of 32
  const int srow = tid >> 3;  // 0..31
  const int sch  = tid & 7;   // 0..7
  const size_t a32 = (size_t)32 * lda;
  const size_t b32 = (size_t)32 * ldb;
  const bf16* gA0 = A + (size_t)srow * lda + sch * 8;
  const bf16* gA1 = gA0 + a32;
  const bf16* gA2 = gA0 + 2 * a32;
  const bf16* gA3 = gA0 + 3 * a32;
  const bf16* gB0 = B + (size_t)(n0 + srow) * ldb + sch * 8;
  const bf16* gB1 = gB0 + b32;
  const bf16* gB2 = gB0 + 2 * b32;
  const bf16* gB3 = gB0 + 3 * b32;
  const bool bv0 = (n0 + srow) < N;
  const bool bv1 = (n0 + srow + 32) < N;
  const bool bv2 = (n0 + srow + 64) < N;
  const bool bv3 = (n0 + srow + 96) < N;

  // LDS drain targets (constant per thread)
  bf16* sA0 = &As[srow][sch * 8];
  bf16* sA1 = &As[srow + 32][sch * 8];
  bf16* sA2 = &As[srow + 64][sch * 8];
  bf16* sA3 = &As[srow + 96][sch * 8];
  bf16* sB0 = &Bs[srow][sch * 8];
  bf16* sB1 = &Bs[srow + 32][sch * 8];
  bf16* sB2 = &Bs[srow + 64][sch * 8];
  bf16* sB3 = &Bs[srow + 96][sch * 8];

  const int4 izero = {0, 0, 0, 0};
  // initial prefetch (kt = 0)
  int4 pa0 = *(const int4*)gA0;
  int4 pa1 = *(const int4*)gA1;
  int4 pa2 = *(const int4*)gA2;
  int4 pa3 = *(const int4*)gA3;
  int4 pb0 = bv0 ? *(const int4*)gB0 : izero;
  int4 pb1 = bv1 ? *(const int4*)gB1 : izero;
  int4 pb2 = bv2 ? *(const int4*)gB2 : izero;
  int4 pb3 = bv3 ? *(const int4*)gB3 : izero;

  const v8f vzero = {0.f, 0.f, 0.f, 0.f, 0.f, 0.f, 0.f, 0.f};
  v8f acc[4][2];
#pragma unroll
  for (int mi = 0; mi < 4; ++mi)
#pragma unroll
    for (int ni = 0; ni < 2; ++ni) acc[mi][ni] = vzero;

  for (int kt = 0; kt < K; kt += BK) {
    // drain prefetch registers into LDS (all b128)
    *(int4*)sA0 = pa0;
    *(int4*)sA1 = pa1;
    *(int4*)sA2 = pa2;
    *(int4*)sA3 = pa3;
    *(int4*)sB0 = pb0;
    *(int4*)sB1 = pb1;
    *(int4*)sB2 = pb2;
    *(int4*)sB3 = pb3;
    __syncthreads();

    if (kt + BK < K) {  // issue next tile's global loads under the WMMAs
      gA0 += BK; gA1 += BK; gA2 += BK; gA3 += BK;
      gB0 += BK; gB1 += BK; gB2 += BK; gB3 += BK;
      pa0 = *(const int4*)gA0;
      pa1 = *(const int4*)gA1;
      pa2 = *(const int4*)gA2;
      pa3 = *(const int4*)gA3;
      pb0 = bv0 ? *(const int4*)gB0 : izero;
      pb1 = bv1 ? *(const int4*)gB1 : izero;
      pb2 = bv2 ? *(const int4*)gB2 : izero;
      pb3 = bv3 ? *(const int4*)gB3 : izero;
    }

    union Frag { v16bf v; v8bf h[2]; };
#pragma unroll
    for (int ks = 0; ks < 2; ++ks) {
      const int kb = ks * 32;
      Frag af[4], bfr[2];
#pragma unroll
      for (int mi = 0; mi < 4; ++mi) {
        int row = wm * 64 + mi * 16 + r;
        af[mi].h[0] = *(const v8bf*)&As[row][kb + g * 8];
        af[mi].h[1] = *(const v8bf*)&As[row][kb + 16 + g * 8];
      }
#pragma unroll
      for (int ni = 0; ni < 2; ++ni) {
        int col = wn * 32 + ni * 16 + r;
        bfr[ni].h[0] = *(const v8bf*)&Bs[col][kb + g * 8];
        bfr[ni].h[1] = *(const v8bf*)&Bs[col][kb + 16 + g * 8];
      }
#pragma unroll
      for (int mi = 0; mi < 4; ++mi)
#pragma unroll
        for (int ni = 0; ni < 2; ++ni)
          acc[mi][ni] = __builtin_amdgcn_wmma_f32_16x16x32_bf16(
              false, af[mi].v, false, bfr[ni].v, (short)0, acc[mi][ni],
              false, false);
    }
    __syncthreads();
  }

  // epilogue: C layout M = e + 8*g, N = r per 16x16 f32 D tiling
#pragma unroll
  for (int mi = 0; mi < 4; ++mi) {
#pragma unroll
    for (int ni = 0; ni < 2; ++ni) {
      const int col = n0 + wn * 32 + ni * 16 + r;
      if (col >= N) continue;
      const float bv = bias ? bias[col] : 0.f;
      const int mbase = m0 + wm * 64 + mi * 16 + g * 8;
      if (CT) {
        union { bf16 b[8]; int4 u; } ob;
#pragma unroll
        for (int e = 0; e < 8; ++e)
          ob.b[e] = (bf16)(acc[mi][ni][e] * scale + bv);
        *(int4*)&Cb[cbase + (size_t)col * ldc + mbase] = ob.u;  // b128 store
      } else {
#pragma unroll
        for (int e = 0; e < 8; ++e) {
          const float x = acc[mi][ni][e] * scale + bv;
          const size_t idx = cbase + (size_t)(mbase + e) * ldc + col;
          if (Cf) Cf[idx] = x;
          if (Cb) Cb[idx] = (bf16)x;
        }
      }
    }
  }
}

// ---------------------------------------------------------------------------
// Row softmax: one block per LEN-wide f32 row, values cached in registers,
// emits bf16 probabilities.
// ---------------------------------------------------------------------------
template <int LEN>
__global__ __launch_bounds__(TPB) void softmax_rows(const float* __restrict__ S,
                                                    bf16* __restrict__ P) {
  constexpr int PER = LEN / TPB;  // 8
  const float* x = S + (size_t)blockIdx.x * LEN;
  bf16* p = P + (size_t)blockIdx.x * LEN;
  __shared__ float red[TPB];
  const int tid = threadIdx.x;

  float xs[PER];
#pragma unroll
  for (int j = 0; j < PER; ++j) xs[j] = x[tid + j * TPB];

  float m = xs[0];
#pragma unroll
  for (int j = 1; j < PER; ++j) m = fmaxf(m, xs[j]);
  red[tid] = m;
  __syncthreads();
  for (int s = TPB / 2; s > 0; s >>= 1) {
    if (tid < s) red[tid] = fmaxf(red[tid], red[tid + s]);
    __syncthreads();
  }
  m = red[0];
  __syncthreads();

  float sum = 0.f;
#pragma unroll
  for (int j = 0; j < PER; ++j) {
    xs[j] = __expf(xs[j] - m);
    sum += xs[j];
  }
  red[tid] = sum;
  __syncthreads();
  for (int s = TPB / 2; s > 0; s >>= 1) {
    if (tid < s) red[tid] += red[tid + s];
    __syncthreads();
  }
  const float inv = 1.f / red[0];
#pragma unroll
  for (int j = 0; j < PER; ++j) p[tid + j * TPB] = (bf16)(xs[j] * inv);
}

// ---------------------------------------------------------------------------
// Host orchestration
// ---------------------------------------------------------------------------
extern "C" void kernel_launch(void* const* d_in, const int* in_sizes, int n_in,
                              void* d_out, int out_size, void* d_ws,
                              size_t ws_size, hipStream_t stream) {
  constexpr int  Bb = 8, S = 2048, D = 1024, Q = 64;
  constexpr long MS = (long)Bb * S;  // 16384 flattened rows

  const float* q   = (const float*)d_in[0];
  const float* k   = (const float*)d_in[1];
  const float* v   = (const float*)d_in[2];
  const float* Wq  = (const float*)d_in[3];
  const float* bq  = (const float*)d_in[4];
  const float* Wk  = (const float*)d_in[5];
  const float* bk  = (const float*)d_in[6];
  const float* Wv  = (const float*)d_in[7];
  const float* bv  = (const float*)d_in[8];
  const float* Wq2 = (const float*)d_in[9];
  const float* bq2 = (const float*)d_in[10];
  const float* Wk2 = (const float*)d_in[11];
  const float* bk2 = (const float*)d_in[12];
  float* out = (float*)d_out;

  size_t off = 0;
  auto alloc = [&](size_t bytes) -> void* {
    void* p = (char*)d_ws + off;
    off += (bytes + 255) & ~(size_t)255;
    return p;
  };
  bf16*  qb    = (bf16*)alloc(MS * D * 2);
  bf16*  kb    = (bf16*)alloc(MS * D * 2);
  bf16*  vb    = (bf16*)alloc(MS * D * 2);
  bf16*  Wqb   = (bf16*)alloc((size_t)D * D * 2);
  bf16*  Wkb   = (bf16*)alloc((size_t)D * D * 2);
  bf16*  Wvb   = (bf16*)alloc((size_t)D * D * 2);
  bf16*  Wq2b  = (bf16*)alloc((size_t)Q * D * 2);
  bf16*  Wk2b  = (bf16*)alloc((size_t)Q * D * 2);
  bf16*  Qpb   = (bf16*)alloc(MS * D * 2);
  bf16*  Kpb   = (bf16*)alloc(MS * D * 2);
  bf16*  VT    = (bf16*)alloc((size_t)D * MS * 2);  // V projection, transposed
  bf16*  Q2b   = (bf16*)alloc(MS * Q * 2);
  bf16*  K2b   = (bf16*)alloc(MS * Q * 2);
  float* score = (float*)alloc((size_t)Bb * S * S * 4);
  bf16*  P     = (bf16*)alloc((size_t)Bb * S * S * 2);
  (void)ws_size; (void)in_sizes; (void)n_in; (void)out_size;

  auto cvt = [&](const float* in, bf16* o, long n) {
    long n4 = n / 4;
    cvt_f32_bf16<<<dim3((unsigned)((n4 + TPB - 1) / TPB)), dim3(TPB), 0,
                   stream>>>(in, o, n4);
  };
  cvt(q, qb, MS * D);
  cvt(k, kb, MS * D);
  cvt(v, vb, MS * D);
  cvt(Wq, Wqb, (long)D * D);
  cvt(Wk, Wkb, (long)D * D);
  cvt(Wv, Wvb, (long)D * D);
  cvt(Wq2, Wq2b, (long)Q * D);
  cvt(Wk2, Wk2b, (long)Q * D);

  auto gemm = [&](bool ct, const bf16* A, const bf16* Bm, const float* bias,
                  float scale, float* Cf, bf16* Cb, int M, int N, int K,
                  int lda, int ldb, int ldc, int batch, long sA, long sB,
                  long sC) {
    dim3 grid((N + 127) / 128, M / 128, batch);
    if (ct)
      gemm_bf16_wmma<true><<<grid, dim3(TPB), 0, stream>>>(
          A, Bm, bias, scale, Cf, Cb, M, N, K, lda, ldb, ldc, sA, sB, sC);
    else
      gemm_bf16_wmma<false><<<grid, dim3(TPB), 0, stream>>>(
          A, Bm, bias, scale, Cf, Cb, M, N, K, lda, ldb, ldc, sA, sB, sC);
  };

  // projections: Xp = X @ W^T + b (bf16 out; V written transposed as VT[D][MS])
  gemm(false, qb, Wqb, bq, 1.f, nullptr, Qpb, (int)MS, D, D, D, D, D, 1, 0, 0, 0);
  gemm(false, kb, Wkb, bk, 1.f, nullptr, Kpb, (int)MS, D, D, D, D, D, 1, 0, 0, 0);
  gemm(true,  vb, Wvb, bv, 1.f, nullptr, VT,  (int)MS, D, D, D, D, (int)MS, 1, 0, 0, 0);
  // low-rank score projections (N=64, zero-padded B tile)
  gemm(false, Qpb, Wq2b, bq2, 1.f, nullptr, Q2b, (int)MS, Q, D, D, D, Q, 1, 0, 0, 0);
  gemm(false, Kpb, Wk2b, bk2, 1.f, nullptr, K2b, (int)MS, Q, D, D, D, Q, 1, 0, 0, 0);
  // score = Q2 @ K2^T / sqrt(D_MODEL), batched (K=64 -> single-stage pipeline)
  gemm(false, Q2b, K2b, nullptr, 0.03125f, score, nullptr, S, S, Q, Q, Q, S,
       Bb, (long)S * Q, (long)S * Q, (long)S * S);
  // softmax over last dim
  softmax_rows<S><<<dim3((unsigned)(Bb * S)), dim3(TPB), 0, stream>>>(score, P);
  // out = P @ V = P @ (VT slice)^T, batched: B rows are VT[d], batch offset
  // is a column offset of b*S into VT's leading dimension MS
  gemm(false, P, VT, nullptr, 1.f, out, nullptr, S, D, S, S, (int)MS, D,
       Bb, (long)S * S, (long)S, (long)S * D);
}
